// MultiScaleMamba_15092515078213
// MI455X (gfx1250) — matrix operand order repaired and checked by reading, not compiled, not run on hardware
//
#include <hip/hip_runtime.h>
#include <hip/hip_bf16.h>

// ---------------------------------------------------------------------------
// Types for CDNA5 WMMA
// ---------------------------------------------------------------------------
typedef _Float16 v16h __attribute__((ext_vector_type(16)));
typedef _Float16 v8h  __attribute__((ext_vector_type(8)));
typedef float    v8f  __attribute__((ext_vector_type(8)));
typedef int      v4i  __attribute__((vector_size(16)));

#define CEILDIV(a, b) (((a) + (b) - 1) / (b))

#define AS1 __attribute__((address_space(1)))
#define AS3 __attribute__((address_space(3)))

#if __has_builtin(__builtin_amdgcn_global_load_async_to_lds_b128) && \
    __has_builtin(__builtin_amdgcn_s_wait_asynccnt)
#define HAVE_ASYNC_LDS 1
#else
#define HAVE_ASYNC_LDS 0
#endif

// ---------------------------------------------------------------------------
// Device helpers
// ---------------------------------------------------------------------------
__device__ __forceinline__ float sigmoidf_(float v) { return 1.f / (1.f + __expf(-v)); }
__device__ __forceinline__ float siluf_(float v)    { return v * sigmoidf_(v); }
__device__ __forceinline__ float softplusf_(float v) {
    if (v > 20.f) return v;
    return log1pf(__expf(v));
}

// ---------------------------------------------------------------------------
// WMMA GEMM:  C[m,n] = sum_k A[m,k] * W[n,k]
//   A: (M x K) f16 row-major,  W: (N x K) f16 row-major (i.e. B^T), C: f32
//   Requires M%16==0, N%16==0, K%32==0, K<=192.
//   One block (4 waves) owns one 16-row A strip, staged once in LDS
//   via GLOBAL_LOAD_ASYNC_TO_LDS_B128 (ASYNCcnt) when available;
//   waves round-robin n-tiles. Fragment layouts per CDNA5 ISA 7.12.2 (wave32).
// ---------------------------------------------------------------------------
__global__ __launch_bounds__(128) void gemm_wmma_f16(
    const _Float16* __restrict__ A, const _Float16* __restrict__ W,
    float* __restrict__ C, int M, int N, int K)
{
    __shared__ __align__(16) _Float16 sA[16 * 192];
    (void)M;
    const int m0   = blockIdx.x << 4;
    const int tid  = threadIdx.x;
    const int lane = tid & 31;
    const int wv   = tid >> 5;

    // ---- stage the 16 x K A-strip into LDS (128-bit granules) ----
    const int nvec = (K << 1);                       // 16*K/8 vec8 elements
    for (int v = tid; v < nvec; v += 128) {
        int idx = v << 3;
        int row = idx / K, col = idx % K;            // K % 8 == 0
#if HAVE_ASYNC_LDS
        __builtin_amdgcn_global_load_async_to_lds_b128(
            (AS1 v4i*)(A + (size_t)(m0 + row) * K + col),
            (AS3 v4i*)(sA + row * K + col), 0, 0);
#else
        *(v8h*)(sA + row * K + col) = *(const v8h*)(A + (size_t)(m0 + row) * K + col);
#endif
    }
#if HAVE_ASYNC_LDS
    __builtin_amdgcn_s_wait_asynccnt(0);
#endif
    __syncthreads();

    const int r  = lane & 15;
    const int hf = lane >> 4;                        // which half-wave
    const _Float16* sarow = sA + r * K;

    for (int n0 = wv << 4; n0 < N; n0 += 64) {       // wave-uniform loop
        const _Float16* wrow = W + (size_t)(n0 + r) * K;
        v8f acc = {0.f, 0.f, 0.f, 0.f, 0.f, 0.f, 0.f, 0.f};
        for (int k0 = 0; k0 < K; k0 += 32) {
            // B 32x16 fragment: lane n=r; lanes 0-15 K {0..7,16..23}, 16-31 {8..15,24..31}
            v8h blo = *(const v8h*)(wrow + k0 + hf * 8);
            v8h bhi = *(const v8h*)(wrow + k0 + hf * 8 + 16);
            // A 16x32 fragment from LDS: lanes 0-15 K {0..7,16..23}, 16-31 {8..15,24..31}
            v8h alo = *(const v8h*)(sarow + k0 + hf * 8);
            v8h ahi = *(const v8h*)(sarow + k0 + hf * 8 + 16);
            if (k0 + 32 < K) __builtin_prefetch(wrow + k0 + 32, 0, 1);
            v16h a, b;
#pragma unroll
            for (int i = 0; i < 8; ++i) {
                a[i] = alo[i]; a[8 + i] = ahi[i];
                b[i] = blo[i]; b[8 + i] = bhi[i];
            }
            acc = __builtin_amdgcn_wmma_f32_16x16x32_f16(
                false, a, false, b, (short)0, acc, false, false);
        }
        // D layout: VGPR i -> row m0 + i + 8*hf, col n0 + r
        float* crow = C + (size_t)(m0 + hf * 8) * N + n0 + r;
#pragma unroll
        for (int i = 0; i < 8; ++i) crow[(size_t)i * N] = acc[i];
    }
}

// ---------------------------------------------------------------------------
// Weight prep: f32 (Ns x Ks) -> f16 (Np x Kp), zero padded
// ---------------------------------------------------------------------------
__global__ void pack_w_f16(const float* __restrict__ src, _Float16* __restrict__ dst,
                           int Ns, int Ks, int Np, int Kp)
{
    int i = blockIdx.x * blockDim.x + threadIdx.x;
    int total = Np * Kp;
    if (i >= total) return;
    int n = i / Kp, k = i % Kp;
    float v = (n < Ns && k < Ks) ? src[(size_t)n * Ks + k] : 0.f;
    dst[i] = (_Float16)v;
}

__global__ void neg_exp_k(const float* __restrict__ a, float* __restrict__ out, int n)
{
    int i = blockIdx.x * blockDim.x + threadIdx.x;
    if (i < n) out[i] = -__expf(a[i]);
}

// ---------------------------------------------------------------------------
// Spectral branch: build A matrix (rows = pixel*4+t, cols d in [0,16) pad 32)
// ---------------------------------------------------------------------------
__global__ void spe_prepare(const float* __restrict__ x, _Float16* __restrict__ A, int npix)
{
    int i = blockIdx.x * blockDim.x + threadIdx.x;
    if (i >= npix * 4 * 32) return;
    int k   = i & 31;
    int row = i >> 5;
    int t   = row & 3;
    int pix = row >> 2;
    int b = pix >> 12, h = (pix >> 6) & 63, w = pix & 63;
    float v = 0.f;
    if (k < 16) v = x[(((size_t)(b * 64 + (t * 16 + k))) << 12) + (h << 6) + w];
    A[i] = (_Float16)v;
}

// Generic (B,C,H,W) f32 -> (pixel, C) f16 (pixel = (b,h,w) flattened)
__global__ void bchw_to_pix_f16(const float* __restrict__ x, _Float16* __restrict__ A,
                                int B_, int C_, int HW)
{
    int i = blockIdx.x * blockDim.x + threadIdx.x;
    int total = B_ * C_ * HW;
    if (i >= total) return;
    int c = i % C_;
    int pix = i / C_;
    int hw = pix % HW;
    int b = pix / HW;
    A[i] = (_Float16)x[((size_t)b * C_ + c) * HW + hw];
}

// ---------------------------------------------------------------------------
// Depthwise causal conv (D_CONV=4) + bias + SiLU
// ---------------------------------------------------------------------------
__global__ void conv_silu_spe(const float* __restrict__ xz, const float* __restrict__ cw,
                              const float* __restrict__ cb, float* __restrict__ xc32,
                              _Float16* __restrict__ xc16, int npix)
{
    int i = blockIdx.x * blockDim.x + threadIdx.x;
    if (i >= npix * 4 * 32) return;
    int e   = i & 31;
    int row = i >> 5;
    int l   = row & 3;
    int pix = row >> 2;
    float s = cb[e];
#pragma unroll
    for (int j = 0; j < 4; ++j) {
        int li = l - 3 + j;
        if (li >= 0) s += cw[e * 4 + j] * xz[(size_t)((pix << 2) + li) * 64 + e];
    }
    float v = siluf_(s);
    xc32[(size_t)row * 32 + e] = v;
    xc16[(size_t)row * 32 + e] = (_Float16)v;
}

__global__ void conv_silu_seq(const float* __restrict__ xz, const float* __restrict__ cw,
                              const float* __restrict__ cb, float* __restrict__ xc32,
                              _Float16* __restrict__ xc16, int L, int D, int XZW)
{
    int i = blockIdx.x * blockDim.x + threadIdx.x;
    if (i >= L * D) return;
    int e = i % D;
    int l = i / D;
    float s = cb[e];
#pragma unroll
    for (int j = 0; j < 4; ++j) {
        int li = l - 3 + j;
        if (li >= 0) s += cw[e * 4 + j] * xz[(size_t)li * XZW + e];
    }
    float v = siluf_(s);
    xc32[(size_t)l * D + e] = v;
    xc16[(size_t)l * D + e] = (_Float16)v;
}

// dt = softplus(xdbl[:, 0:4] @ dt_proj_w^T + b)   (spatial: dt_rank=4, xdbl ld=48)
__global__ void dt_softplus(const float* __restrict__ xdbl, const float* __restrict__ dtw,
                            const float* __restrict__ dtb, float* __restrict__ dt, int L)
{
    int i = blockIdx.x * blockDim.x + threadIdx.x;
    if (i >= L * 128) return;
    int e = i & 127;
    int l = i >> 7;
    float s = dtb[e];
#pragma unroll
    for (int r = 0; r < 4; ++r) s += xdbl[(size_t)l * 48 + r] * dtw[e * 4 + r];
    dt[i] = softplusf_(s);
}

// ---------------------------------------------------------------------------
// Spectral scan: per-(pixel, e), L=4, 16 states in registers. dt_rank=1.
// xdbl cols: [0]=dt_raw, [1..16]=B, [17..32]=C (ld=48)
// ---------------------------------------------------------------------------
__global__ void spe_scan(const float* __restrict__ xz, const float* __restrict__ xc32,
                         const float* __restrict__ xdbl, const float* __restrict__ dtw,
                         const float* __restrict__ dtb, const float* __restrict__ aneg,
                         const float* __restrict__ Dskip, _Float16* __restrict__ y16, int npix)
{
    int i = blockIdx.x * blockDim.x + threadIdx.x;
    if (i >= npix * 32) return;
    int e   = i & 31;
    int pix = i >> 5;
    float h[16];
    float an[16];
#pragma unroll
    for (int n = 0; n < 16; ++n) { h[n] = 0.f; an[n] = aneg[e * 16 + n]; }
    float dw = dtw[e], db = dtb[e], Dv = Dskip[e];
    for (int l = 0; l < 4; ++l) {
        size_t row = (size_t)pix * 4 + l;
        const float* xd = xdbl + row * 48;
        float dtv = softplusf_(xd[0] * dw + db);
        float xcv = xc32[row * 32 + e];
        float y = 0.f;
#pragma unroll
        for (int n = 0; n < 16; ++n) {
            float a = __expf(dtv * an[n]);
            h[n] = h[n] * a + dtv * xd[1 + n] * xcv;
            y += h[n] * xd[17 + n];
        }
        float z = xz[row * 64 + 32 + e];
        y16[row * 32 + e] = (_Float16)((y + xcv * Dv) * siluf_(z));
    }
}

// ---------------------------------------------------------------------------
// Spatial/s4 chunked parallel scan over linear recurrence h = a*h + b.
// 2048 (e,n) pairs; thread t owns e=t>>2, n in [4*(t&3), 4*(t&3)+4).
// Pass1: per chunk compute P=prod(a), S=scan end with h0=0.
// Pass2: sequential prefix over chunks (one block).
// Pass3: re-scan with injected initial state; y[l,e]=sum_n h*C reduced via shfl.
// xdbl cols: [0..3]=dt_raw, [4..19]=B, [20..35]=C (ld=48)
// ---------------------------------------------------------------------------
__global__ __launch_bounds__(512) void scan_pass1(
    const float* __restrict__ dt, const float* __restrict__ xc,
    const float* __restrict__ xdbl, const float* __restrict__ aneg,
    float* __restrict__ P, float* __restrict__ S, int chunk_len)
{
    int c = blockIdx.x, t = threadIdx.x;
    int e = t >> 2, n0 = (t & 3) * 4;
    float h[4] = {0.f, 0.f, 0.f, 0.f};
    float p[4] = {1.f, 1.f, 1.f, 1.f};
    float an[4];
#pragma unroll
    for (int j = 0; j < 4; ++j) an[j] = aneg[e * 16 + n0 + j];
    int l0 = c * chunk_len;
    for (int l = l0; l < l0 + chunk_len; ++l) {
        float dtv = dt[(size_t)l * 128 + e];
        float xcv = xc[(size_t)l * 128 + e];
        const float* xd = xdbl + (size_t)l * 48 + 4;
#pragma unroll
        for (int j = 0; j < 4; ++j) {
            float a = __expf(dtv * an[j]);
            h[j] = h[j] * a + dtv * xd[n0 + j] * xcv;
            p[j] *= a;
        }
    }
    size_t base = (size_t)c * 2048 + t * 4;
#pragma unroll
    for (int j = 0; j < 4; ++j) { P[base + j] = p[j]; S[base + j] = h[j]; }
}

__global__ __launch_bounds__(512) void scan_pass2(
    const float* __restrict__ P, const float* __restrict__ S,
    float* __restrict__ Init, int nchunks)
{
    int t = threadIdx.x;
    float carry[4] = {0.f, 0.f, 0.f, 0.f};
    for (int c = 0; c < nchunks; ++c) {
        size_t base = (size_t)c * 2048 + t * 4;
#pragma unroll
        for (int j = 0; j < 4; ++j) {
            Init[base + j] = carry[j];
            carry[j] = P[base + j] * carry[j] + S[base + j];
        }
    }
}

__global__ __launch_bounds__(512) void scan_pass3(
    const float* __restrict__ dt, const float* __restrict__ xc,
    const float* __restrict__ xdbl, const float* __restrict__ xz,
    const float* __restrict__ aneg, const float* __restrict__ Init,
    const float* __restrict__ Dskip, _Float16* __restrict__ y16, int chunk_len)
{
    int c = blockIdx.x, t = threadIdx.x;
    int e = t >> 2, n0 = (t & 3) * 4;
    float h[4], an[4];
    size_t base = (size_t)c * 2048 + t * 4;
#pragma unroll
    for (int j = 0; j < 4; ++j) { h[j] = Init[base + j]; an[j] = aneg[e * 16 + n0 + j]; }
    float Dv = Dskip[e];
    int l0 = c * chunk_len;
    for (int l = l0; l < l0 + chunk_len; ++l) {
        float dtv = dt[(size_t)l * 128 + e];
        float xcv = xc[(size_t)l * 128 + e];
        const float* xd = xdbl + (size_t)l * 48;
        float part = 0.f;
#pragma unroll
        for (int j = 0; j < 4; ++j) {
            float a = __expf(dtv * an[j]);
            h[j] = h[j] * a + dtv * xd[4 + n0 + j] * xcv;
            part += h[j] * xd[20 + n0 + j];
        }
        part += __shfl_xor(part, 1, 32);
        part += __shfl_xor(part, 2, 32);
        if ((t & 3) == 0) {
            float z = xz[(size_t)l * 256 + 128 + e];
            y16[(size_t)l * 128 + e] = (_Float16)((part + xcv * Dv) * siluf_(z));
        }
    }
}

// ---------------------------------------------------------------------------
// Reformat mamba outputs back to (B,C,H,W)
// ---------------------------------------------------------------------------
__global__ void pix_to_bchw(const float* __restrict__ src, float* __restrict__ dst,
                            int B_, int C_, int HW)
{
    int i = blockIdx.x * blockDim.x + threadIdx.x;
    int total = B_ * C_ * HW;
    if (i >= total) return;
    int hw = i % HW;
    int c  = (i / HW) % C_;
    int b  = i / (HW * C_);
    dst[i] = src[((size_t)b * HW + hw) * C_ + c];
}

__global__ void spe_to_bchw(const float* __restrict__ src, float* __restrict__ dst)
{
    int i = blockIdx.x * blockDim.x + threadIdx.x;
    if (i >= 1048576) return;
    int hw = i & 4095;
    int c  = (i >> 12) & 63;
    int b  = i >> 18;
    int pix = (b << 12) + hw;
    int t = c >> 4, d = c & 15;
    dst[i] = src[((size_t)pix * 4 + t) * 16 + d];
}

// ---------------------------------------------------------------------------
// GroupNorm (two pass): stats per (b,g) via LDS reduction, then apply+silu+res
// ---------------------------------------------------------------------------
__global__ __launch_bounds__(256) void gn_stats(const float* __restrict__ y,
                                                float* __restrict__ stats,
                                                int C, int HW, int groups)
{
    int bg = blockIdx.x;
    int g = bg % groups, b = bg / groups;
    int Cg = C / groups;
    size_t base = ((size_t)b * C + (size_t)g * Cg) * HW;
    int n = Cg * HW;
    float s = 0.f, s2 = 0.f;
    for (int i = threadIdx.x; i < n; i += blockDim.x) {
        float v = y[base + i];
        s += v; s2 += v * v;
    }
    __shared__ float sh[256], sh2[256];
    sh[threadIdx.x] = s; sh2[threadIdx.x] = s2;
    __syncthreads();
    for (int o = 128; o > 0; o >>= 1) {
        if (threadIdx.x < o) {
            sh[threadIdx.x]  += sh[threadIdx.x + o];
            sh2[threadIdx.x] += sh2[threadIdx.x + o];
        }
        __syncthreads();
    }
    if (threadIdx.x == 0) {
        float mu  = sh[0] / n;
        float var = sh2[0] / n - mu * mu;
        stats[bg * 2]     = mu;
        stats[bg * 2 + 1] = rsqrtf(var + 1e-5f);
    }
}

__global__ void gn_apply_silu_res(const float* __restrict__ y, const float* __restrict__ resid,
                                  const float* __restrict__ stats, const float* __restrict__ gamma,
                                  const float* __restrict__ beta, float* __restrict__ out,
                                  int B_, int C, int HW, int groups)
{
    int i = blockIdx.x * blockDim.x + threadIdx.x;
    int total = B_ * C * HW;
    if (i >= total) return;
    int c = (i / HW) % C;
    int b = i / (HW * C);
    int g = c / (C / groups);
    float mu = stats[(b * groups + g) * 2];
    float is = stats[(b * groups + g) * 2 + 1];
    float xn = (y[i] - mu) * is * gamma[c] + beta[c];
    out[i] = resid[i] + siluf_(xn);
}

// ---------------------------------------------------------------------------
// avg_pool 4x4 and align-corners bilinear upsample 16->64
// ---------------------------------------------------------------------------
__global__ void avg_pool4(const float* __restrict__ x, float* __restrict__ xp)
{
    int i = blockIdx.x * blockDim.x + threadIdx.x;
    if (i >= 4 * 64 * 16 * 16) return;
    int w2 = i & 15, h2 = (i >> 4) & 15, c = (i >> 8) & 63, b = i >> 14;
    float s = 0.f;
#pragma unroll
    for (int dy = 0; dy < 4; ++dy)
#pragma unroll
        for (int dx = 0; dx < 4; ++dx)
            s += x[(((size_t)(b * 64 + c)) << 12) + (h2 * 4 + dy) * 64 + (w2 * 4 + dx)];
    xp[i] = s * (1.f / 16.f);
}

__global__ void upsample_ac(const float* __restrict__ s, float* __restrict__ d)
{
    int i = blockIdx.x * blockDim.x + threadIdx.x;
    if (i >= 1048576) return;
    int w = i & 63, h = (i >> 6) & 63, c = (i >> 12) & 63, b = i >> 18;
    const float sc = 15.f / 63.f;
    float fy = h * sc, fx = w * sc;
    int y0 = (int)fy, x0 = (int)fx;
    int y1 = y0 + 1 < 15 ? y0 + 1 : 15;
    int x1 = x0 + 1 < 15 ? x0 + 1 : 15;
    float wy = fy - y0, wx = fx - x0;
    const float* p = s + ((size_t)(b * 64 + c)) * 256;
    float a  = p[y0 * 16 + x0], bb = p[y0 * 16 + x1];
    float cc = p[y1 * 16 + x0], dd = p[y1 * 16 + x1];
    d[i] = (a * (1.f - wx) + bb * wx) * (1.f - wy) + (cc * (1.f - wx) + dd * wx) * wy;
}

// ---------------------------------------------------------------------------
// Head: concat feats -> f16, sigmoid-gated fuse, final proj bias + reshape
// ---------------------------------------------------------------------------
__global__ void build_feats(const float* __restrict__ spe, const float* __restrict__ spa,
                            const float* __restrict__ s4, _Float16* __restrict__ f)
{
    int i = blockIdx.x * blockDim.x + threadIdx.x;
    if (i >= 16384 * 192) return;
    int c = i % 192;
    int pix = i / 192;
    int br = c / 64, cc = c % 64;
    int hw = pix & 4095, b = pix >> 12;
    size_t idx = (((size_t)(b * 64 + cc)) << 12) + hw;
    float v = (br == 0) ? spe[idx] : (br == 1) ? spa[idx] : s4[idx];
    f[i] = (_Float16)v;
}

__global__ void fuse_attn(const float* __restrict__ attn, const float* __restrict__ ab,
                          const float* __restrict__ spe, const float* __restrict__ spa,
                          const float* __restrict__ s4, _Float16* __restrict__ fused)
{
    int i = blockIdx.x * blockDim.x + threadIdx.x;
    if (i >= 16384 * 64) return;
    int c = i & 63;
    int pix = i >> 6;
    int hw = pix & 4095, b = pix >> 12;
    size_t idx = (((size_t)(b * 64 + c)) << 12) + hw;
    const float* ar = attn + (size_t)pix * 192;
    float a0 = sigmoidf_(ar[c] + ab[c]);
    float a1 = sigmoidf_(ar[64 + c] + ab[64 + c]);
    float a2 = sigmoidf_(ar[128 + c] + ab[128 + c]);
    fused[i] = (_Float16)(spe[idx] * a0 + spa[idx] * a1 + s4[idx] * a2);
}

__global__ void proj_bias_to_bchw(const float* __restrict__ p, const float* __restrict__ pb,
                                  float* __restrict__ dst)
{
    int i = blockIdx.x * blockDim.x + threadIdx.x;
    if (i >= 1048576) return;
    int hw = i & 4095;
    int c  = (i >> 12) & 63;
    int b  = i >> 18;
    int pix = (b << 12) + hw;
    dst[i] = p[(size_t)pix * 64 + c] + pb[c];
}

// ---------------------------------------------------------------------------
// Host orchestration
// ---------------------------------------------------------------------------
extern "C" void kernel_launch(void* const* d_in, const int* in_sizes, int n_in,
                              void* d_out, int out_size, void* d_ws, size_t ws_size,
                              hipStream_t stream)
{
    (void)in_sizes; (void)n_in; (void)out_size; (void)ws_size;

    const float* x = (const float*)d_in[0];
    // spe params (d_model=16, d_inner=32, dt_rank=1)
    const float* spe_inw = (const float*)d_in[1];
    const float* spe_cw  = (const float*)d_in[2];
    const float* spe_cb  = (const float*)d_in[3];
    const float* spe_xpw = (const float*)d_in[4];
    const float* spe_dtw = (const float*)d_in[5];
    const float* spe_dtb = (const float*)d_in[6];
    const float* spe_alog= (const float*)d_in[7];
    const float* spe_D   = (const float*)d_in[8];
    const float* spe_ow  = (const float*)d_in[9];
    const float* spe_gg  = (const float*)d_in[10];
    const float* spe_gb  = (const float*)d_in[11];
    // spa params (d_model=64, d_inner=128, dt_rank=4)
    const float* spa_inw = (const float*)d_in[12];
    const float* spa_cw  = (const float*)d_in[13];
    const float* spa_cb  = (const float*)d_in[14];
    const float* spa_xpw = (const float*)d_in[15];
    const float* spa_dtw = (const float*)d_in[16];
    const float* spa_dtb = (const float*)d_in[17];
    const float* spa_alog= (const float*)d_in[18];
    const float* spa_D   = (const float*)d_in[19];
    const float* spa_ow  = (const float*)d_in[20];
    const float* spa_gg  = (const float*)d_in[21];
    const float* spa_gb  = (const float*)d_in[22];
    // s4 params
    const float* s4_inw  = (const float*)d_in[23];
    const float* s4_cw   = (const float*)d_in[24];
    const float* s4_cb   = (const float*)d_in[25];
    const float* s4_xpw  = (const float*)d_in[26];
    const float* s4_dtw  = (const float*)d_in[27];
    const float* s4_dtb  = (const float*)d_in[28];
    const float* s4_alog = (const float*)d_in[29];
    const float* s4_D    = (const float*)d_in[30];
    const float* s4_ow   = (const float*)d_in[31];
    const float* s4_gg   = (const float*)d_in[32];
    const float* s4_gb   = (const float*)d_in[33];
    // head
    const float* attn_w  = (const float*)d_in[34];
    const float* attn_b  = (const float*)d_in[35];
    const float* proj_w  = (const float*)d_in[36];
    const float* proj_b  = (const float*)d_in[37];
    const float* hd_gg   = (const float*)d_in[38];
    const float* hd_gb   = (const float*)d_in[39];

    // ---- workspace carve (deterministic) ----
    size_t off = 0;
    char* wsb = (char*)d_ws;
    auto carve = [&](size_t bytes) -> char* {
        char* p = wsb + off;
        off += (bytes + 255) & ~(size_t)255;
        return p;
    };
    float* spe_out  = (float*)carve(1048576 * 4);
    float* spa_out  = (float*)carve(1048576 * 4);
    float* s4_out   = (float*)carve(1048576 * 4);
    float* xp       = (float*)carve(65536 * 4);
    float* s4s_out  = (float*)carve(65536 * 4);
    _Float16* w_spe_in  = (_Float16*)carve(64 * 32 * 2);
    _Float16* w_spe_xp  = (_Float16*)carve(48 * 32 * 2);
    _Float16* w_spe_out = (_Float16*)carve(16 * 32 * 2);
    _Float16* w_spa_in  = (_Float16*)carve(256 * 64 * 2);
    _Float16* w_spa_xp  = (_Float16*)carve(48 * 128 * 2);
    _Float16* w_spa_out = (_Float16*)carve(64 * 128 * 2);
    _Float16* w_s4_in   = (_Float16*)carve(256 * 64 * 2);
    _Float16* w_s4_xp   = (_Float16*)carve(48 * 128 * 2);
    _Float16* w_s4_out  = (_Float16*)carve(64 * 128 * 2);
    _Float16* w_attn    = (_Float16*)carve(192 * 192 * 2);
    _Float16* w_proj    = (_Float16*)carve(64 * 64 * 2);
    float* an_spe = (float*)carve(512 * 4);
    float* an_spa = (float*)carve(2048 * 4);
    float* an_s4  = (float*)carve(2048 * 4);
    float* stats  = (float*)carve(64 * 4);
    _Float16* A16   = (_Float16*)carve(2097152 * 2);
    float*    xz    = (float*)   carve(4194304 * 4);
    float*    xc32  = (float*)   carve(2097152 * 4);
    _Float16* xc16  = (_Float16*)carve(2097152 * 2);
    float*    xdbl  = (float*)   carve(3145728 * 4);
    float*    dtbuf = (float*)   carve(2097152 * 4);
    _Float16* y16   = (_Float16*)carve(2097152 * 2);
    float*    ymb   = (float*)   carve(1048576 * 4);
    float*    ybchw = (float*)   carve(1048576 * 4);
    float*    scanP = (float*)   carve(131072 * 4);
    float*    scanS = (float*)   carve(131072 * 4);
    float*    scanI = (float*)   carve(131072 * 4);
    _Float16* feats16 = (_Float16*)carve(3145728 * 2);
    float*    attn32  = (float*)   carve(3145728 * 4);
    _Float16* fused16 = (_Float16*)carve(1048576 * 2);

    auto gemm = [&](const _Float16* A, const _Float16* W, float* C, int M, int N, int K) {
        gemm_wmma_f16<<<M / 16, 128, 0, stream>>>(A, W, C, M, N, K);
    };
    auto ew = [&](int n) { return dim3(CEILDIV(n, 256)); };

    // ---- weight prep ----
    pack_w_f16<<<ew(64 * 32), 256, 0, stream>>>(spe_inw, w_spe_in, 64, 16, 64, 32);
    pack_w_f16<<<ew(48 * 32), 256, 0, stream>>>(spe_xpw, w_spe_xp, 33, 32, 48, 32);
    pack_w_f16<<<ew(16 * 32), 256, 0, stream>>>(spe_ow, w_spe_out, 16, 32, 16, 32);
    pack_w_f16<<<ew(256 * 64), 256, 0, stream>>>(spa_inw, w_spa_in, 256, 64, 256, 64);
    pack_w_f16<<<ew(48 * 128), 256, 0, stream>>>(spa_xpw, w_spa_xp, 36, 128, 48, 128);
    pack_w_f16<<<ew(64 * 128), 256, 0, stream>>>(spa_ow, w_spa_out, 64, 128, 64, 128);
    pack_w_f16<<<ew(256 * 64), 256, 0, stream>>>(s4_inw, w_s4_in, 256, 64, 256, 64);
    pack_w_f16<<<ew(48 * 128), 256, 0, stream>>>(s4_xpw, w_s4_xp, 36, 128, 48, 128);
    pack_w_f16<<<ew(64 * 128), 256, 0, stream>>>(s4_ow, w_s4_out, 64, 128, 64, 128);
    pack_w_f16<<<ew(192 * 192), 256, 0, stream>>>(attn_w, w_attn, 192, 192, 192, 192);
    pack_w_f16<<<ew(64 * 64), 256, 0, stream>>>(proj_w, w_proj, 64, 64, 64, 64);
    neg_exp_k<<<ew(512), 256, 0, stream>>>(spe_alog, an_spe, 512);
    neg_exp_k<<<ew(2048), 256, 0, stream>>>(spa_alog, an_spa, 2048);
    neg_exp_k<<<ew(2048), 256, 0, stream>>>(s4_alog, an_s4, 2048);

    // ================= spectral branch (16384 seqs, L=4, d=16) =================
    spe_prepare<<<ew(65536 * 32), 256, 0, stream>>>(x, A16, 16384);
    gemm(A16, w_spe_in, xz, 65536, 64, 32);
    conv_silu_spe<<<ew(65536 * 32), 256, 0, stream>>>(xz, spe_cw, spe_cb, xc32, xc16, 16384);
    gemm(xc16, w_spe_xp, xdbl, 65536, 48, 32);
    spe_scan<<<ew(16384 * 32), 256, 0, stream>>>(xz, xc32, xdbl, spe_dtw, spe_dtb,
                                                 an_spe, spe_D, y16, 16384);
    gemm(y16, w_spe_out, ymb, 65536, 16, 32);
    spe_to_bchw<<<ew(1048576), 256, 0, stream>>>(ymb, ybchw);
    gn_stats<<<16, 256, 0, stream>>>(ybchw, stats, 64, 4096, 4);
    gn_apply_silu_res<<<ew(1048576), 256, 0, stream>>>(ybchw, x, stats, spe_gg, spe_gb,
                                                       spe_out, 4, 64, 4096, 4);

    // ================= spatial branch (1 seq, L=16384, d=64) =================
    bchw_to_pix_f16<<<ew(1048576), 256, 0, stream>>>(x, A16, 4, 64, 4096);
    gemm(A16, w_spa_in, xz, 16384, 256, 64);
    conv_silu_seq<<<ew(16384 * 128), 256, 0, stream>>>(xz, spa_cw, spa_cb, xc32, xc16,
                                                       16384, 128, 256);
    gemm(xc16, w_spa_xp, xdbl, 16384, 48, 128);
    dt_softplus<<<ew(16384 * 128), 256, 0, stream>>>(xdbl, spa_dtw, spa_dtb, dtbuf, 16384);
    scan_pass1<<<64, 512, 0, stream>>>(dtbuf, xc32, xdbl, an_spa, scanP, scanS, 256);
    scan_pass2<<<1, 512, 0, stream>>>(scanP, scanS, scanI, 64);
    scan_pass3<<<64, 512, 0, stream>>>(dtbuf, xc32, xdbl, xz, an_spa, scanI, spa_D, y16, 256);
    gemm(y16, w_spa_out, ymb, 16384, 64, 128);
    pix_to_bchw<<<ew(1048576), 256, 0, stream>>>(ymb, ybchw, 4, 64, 4096);
    gn_stats<<<16, 256, 0, stream>>>(ybchw, stats, 64, 4096, 4);
    gn_apply_silu_res<<<ew(1048576), 256, 0, stream>>>(ybchw, x, stats, spa_gg, spa_gb,
                                                       spa_out, 4, 64, 4096, 4);

    // ================= s4 branch (pooled, L=1024) =================
    avg_pool4<<<ew(65536), 256, 0, stream>>>(x, xp);
    bchw_to_pix_f16<<<ew(65536), 256, 0, stream>>>(xp, A16, 4, 64, 256);
    gemm(A16, w_s4_in, xz, 1024, 256, 64);
    conv_silu_seq<<<ew(1024 * 128), 256, 0, stream>>>(xz, s4_cw, s4_cb, xc32, xc16,
                                                      1024, 128, 256);
    gemm(xc16, w_s4_xp, xdbl, 1024, 48, 128);
    dt_softplus<<<ew(1024 * 128), 256, 0, stream>>>(xdbl, s4_dtw, s4_dtb, dtbuf, 1024);
    scan_pass1<<<4, 512, 0, stream>>>(dtbuf, xc32, xdbl, an_s4, scanP, scanS, 256);
    scan_pass2<<<1, 512, 0, stream>>>(scanP, scanS, scanI, 4);
    scan_pass3<<<4, 512, 0, stream>>>(dtbuf, xc32, xdbl, xz, an_s4, scanI, s4_D, y16, 256);
    gemm(y16, w_s4_out, ymb, 1024, 64, 128);
    pix_to_bchw<<<ew(65536), 256, 0, stream>>>(ymb, ybchw, 4, 64, 256);
    gn_stats<<<16, 256, 0, stream>>>(ybchw, stats, 64, 256, 4);
    gn_apply_silu_res<<<ew(65536), 256, 0, stream>>>(ybchw, xp, stats, s4_gg, s4_gb,
                                                     s4s_out, 4, 64, 256, 4);
    upsample_ac<<<ew(1048576), 256, 0, stream>>>(s4s_out, s4_out);

    // ================= fusion head =================
    build_feats<<<ew(16384 * 192), 256, 0, stream>>>(spe_out, spa_out, s4_out, feats16);
    gemm(feats16, w_attn, attn32, 16384, 192, 192);
    fuse_attn<<<ew(16384 * 64), 256, 0, stream>>>(attn32, attn_b, spe_out, spa_out,
                                                  s4_out, fused16);
    gemm(fused16, w_proj, ymb, 16384, 64, 64);
    proj_bias_to_bchw<<<ew(1048576), 256, 0, stream>>>(ymb, proj_b, ybchw);
    gn_stats<<<16, 256, 0, stream>>>(ybchw, stats, 64, 4096, 4);
    gn_apply_silu_res<<<ew(1048576), 256, 0, stream>>>(ybchw, x, stats, hd_gg, hd_gb,
                                                       (float*)d_out, 4, 64, 4096, 4);
}
